// Net_44710609551888
// MI455X (gfx1250) — matrix operand order repaired and checked by reading, compiled, not run on hardware
//
#include <hip/hip_runtime.h>

typedef _Float16 h16;
typedef _Float16 v16h __attribute__((ext_vector_type(16)));
typedef _Float16 v8h  __attribute__((ext_vector_type(8)));
typedef float    v8f  __attribute__((ext_vector_type(8)));
typedef unsigned int u32x4 __attribute__((ext_vector_type(4)));
typedef int          i32x8 __attribute__((ext_vector_type(8)));
typedef int          i32x4 __attribute__((ext_vector_type(4)));

union Frag16 { v16h v; v8h h[2]; };

#define NPTS 32768
#define PPC  2048
#define KNNK 3
#define NEDGE (NPTS * KNNK)
#define BN_INV 0.99999500003749968f
#define BIGF 3.0e38f

// ---------------------------------------------------------------- TDM: 2D tile (rows x width f32) global -> LDS
// D# bitfields per CDNA5 ISA ch.8 (group0: count/lds_addr/global_addr/type=2;
// group1: data_size=4B, tensor_dim0/1, tile_dim0/1, dim0 stride).
__device__ __forceinline__ void tdm_load_2d_f32(unsigned int lds_off, const float* gsrc,
                                                int width, int rows, int stride) {
  unsigned long long ga = (unsigned long long)(uintptr_t)gsrc;
  u32x4 g0;
  g0[0] = 1u;                                    // count = 1 (valid descriptor)
  g0[1] = lds_off;                               // lds_addr (bytes)
  g0[2] = (unsigned int)ga;                      // global_addr[31:0]
  g0[3] = (unsigned int)(ga >> 32) | (2u << 30); // global_addr[56:32] | type=2
  i32x8 g1;
  g1[0] = (2 << 16);                                      // wg_mask=0 | data_size=2 (4B)
  g1[1] = (width & 0xffff) << 16;                         // tensor_dim0 lo16
  g1[2] = ((width >> 16) & 0xffff) | ((rows & 0xffff) << 16);   // dim0 hi | dim1 lo
  g1[3] = ((rows >> 16) & 0xffff) | ((width & 0xffff) << 16);   // dim1 hi | tile_dim0
  g1[4] = (rows & 0xffff);                                // tile_dim1 | tile_dim2=0
  g1[5] = stride;                                         // tensor_dim0_stride lo32
  g1[6] = 0;                                              // stride hi | dim1_stride lo
  g1[7] = 0;
  i32x4 g2;
  g2[0] = 1;          // tensor_dim2 = 1
  g2[1] = 1;          // tensor_dim3 = 1
  g2[2] = 0;          // tensor_dim2_stride lo
  g2[3] = 0;          // stride hi | tile_dim3 = 0
  i32x4 g3;
  g3[0] = 0;          // tensor_dim3_stride lo
  g3[1] = (1 << 16);  // stride hi | tensor_dim4 = 1
  g3[2] = 0;          // dim4 hi | tile_dim4 = 0
  g3[3] = 0;
#if __clang_major__ >= 23
  i32x8 gz = {};
  __builtin_amdgcn_tensor_load_to_lds(g0, g1, g2, g3, gz, 0);
#else
  __builtin_amdgcn_tensor_load_to_lds(g0, g1, g2, g3, 0);
#endif
}

// ---------------------------------------------------------------- utilities
__global__ void build_x0(const float* __restrict__ x, const float* __restrict__ pos,
                         float* __restrict__ x0f, h16* __restrict__ x0h) {
  int t = blockIdx.x * blockDim.x + threadIdx.x;
  if (t >= NPTS * 6) return;
  int i = t / 6, d = t % 6;
  float v = (d < 3) ? x[i * 3 + d] : pos[i * 3 + (d - 3)];
  x0f[t] = v;
  x0h[t] = (h16)v;
}

// f32 weight [rows_in, cols] -> f16 WMMA B-fragment layout, rows zero-padded to rows_pad.
// Fragment (kchunk, ctile) = 32x16 halves; lane = col%16 + 16*(kin/16); half j <-> kin%16.
// Per-lane 16 halves stored contiguously: dst = (frag*32 + lane)*16 + j.
__global__ void convert_swizzle_b(const float* __restrict__ in, h16* __restrict__ out,
                                  int rows_in, int rows_pad, int cols) {
  int t = blockIdx.x * blockDim.x + threadIdx.x;
  if (t >= rows_pad * cols) return;
  int k = t / cols, c = t % cols;
  float v = (k < rows_in) ? in[k * cols + c] : 0.0f;
  int kchunk = k >> 5, kin = k & 31;
  int ct = c >> 4, cl = c & 15;
  int lane = cl + ((kin >> 4) << 4);
  int j = kin & 15;
  size_t dst = ((size_t)(kchunk * (cols >> 4) + ct) * 32 + lane) * 16 + j;
  out[dst] = (h16)v;
}

template <int D>
__global__ void row_norms(const float* __restrict__ f, float* __restrict__ sq) {
  int i = blockIdx.x * blockDim.x + threadIdx.x;
  if (i >= NPTS) return;
  float s = 0.0f;
#pragma unroll
  for (int d = 0; d < D; ++d) { float v = f[i * D + d]; s = fmaf(v, v, s); }
  sq[i] = s;
}

// ---------------------------------------------------------------- kNN (top-3 smallest dist, ascending-index tiebreak, includes self)
// Candidate tile staged into LDS by the Tensor Data Mover (wave 0 issues, waits TENSORcnt).
template <int D>
__global__ __launch_bounds__(256) void knn_kernel(const float* __restrict__ feat,
                                                  const float* __restrict__ sq,
                                                  int* __restrict__ nbr) {
  const int TILE = 128;
  __shared__ float cf[TILE * D];
  __shared__ float cs[TILE];
  int tid = threadIdx.x;
  int q = blockIdx.x * 256 + tid;      // 2048 % 256 == 0 -> block stays in one cloud
  int cloud = q / PPC;
  int base = cloud * PPC;

  float qf[D];
#pragma unroll
  for (int d = 0; d < D; ++d) qf[d] = feat[(size_t)q * D + d];
  float qs = sq[q];

  float d0 = BIGF, d1 = BIGF, d2 = BIGF;
  int i0 = base, i1 = base, i2 = base;

  unsigned int cf_lds = (unsigned int)(uintptr_t)&cf[0];

  for (int t0 = 0; t0 < PPC; t0 += TILE) {
    __syncthreads();
    if ((tid >> 5) == 0) {   // wave 0 drives the TDM
      tdm_load_2d_f32(cf_lds, feat + (size_t)(base + t0) * D, D, TILE, D);
      __builtin_amdgcn_s_wait_tensorcnt((short)0);
    }
    for (int k = tid; k < TILE; k += 256)
      cs[k] = sq[base + t0 + k];
    __syncthreads();
    for (int c = 0; c < TILE; ++c) {
      float dot = 0.0f;
#pragma unroll
      for (int d = 0; d < D; ++d) dot = fmaf(qf[d], cf[c * D + d], dot);
      float dist = qs + cs[c] - 2.0f * dot;
      int j = base + t0 + c;
      if (dist < d2) {
        if (dist < d1) {
          if (dist < d0) { d2 = d1; i2 = i1; d1 = d0; i1 = i0; d0 = dist; i0 = j; }
          else           { d2 = d1; i2 = i1; d1 = dist; i1 = j; }
        } else           { d2 = dist; i2 = j; }
      }
    }
  }
  nbr[q * 3 + 0] = i0;
  nbr[q * 3 + 1] = i1;
  nbr[q * 3 + 2] = i2;
}

// ---------------------------------------------------------------- edge feature gather: [x_i, x_j - x_i], K-padded
template <int D, int DPAD>
__global__ void edge_gather(const h16* __restrict__ xh, const int* __restrict__ nbr,
                            h16* __restrict__ E) {
  int e = blockIdx.x * blockDim.x + threadIdx.x;
  if (e >= NEDGE) return;
  int i = e / KNNK;
  int j = nbr[e];
  h16* er = E + (size_t)e * DPAD;
#pragma unroll
  for (int d = 0; d < D; ++d) {
    float xi = (float)xh[(size_t)i * D + d];
    float xj = (float)xh[(size_t)j * D + d];
    er[d]     = (h16)xi;
    er[D + d] = (h16)(xj - xi);
  }
#pragma unroll
  for (int d = 2 * D; d < DPAD; ++d) er[d] = (h16)0.0f;
}

// ---------------------------------------------------------------- WMMA GEMM + ReLU + BN-affine epilogue
// out[M, DOUT] = bn(relu(A[M, DIN] @ W + bias)); A row-major f16, W pre-swizzled to
// B-fragments, f32 accum. Block = 8 waves; each wave: 16 rows x 64 cols (4 WMMA tiles).
template <int DIN, int DOUT>
__global__ __launch_bounds__(256) void gemm_bn_relu(
    const h16* __restrict__ A, const h16* __restrict__ Wsw,
    const float* __restrict__ bias, const float* __restrict__ gamma,
    const float* __restrict__ beta, h16* __restrict__ out) {
  const int lane = threadIdx.x & 31;
  const int wave = threadIdx.x >> 5;
  const int r0   = blockIdx.x * 128 + wave * 16;
  const int cg   = blockIdx.y * 64;
  const int arow = r0 + (lane & 15);
  const int hi   = lane >> 4;
  const int ahalf = hi ? 16 : 0;  // A f16 layout: each half-wave holds 16 contiguous K
  const int ctile0 = cg >> 4;

  v8f acc[4] = {};

  for (int kc = 0; kc < DIN; kc += 32) {
    const h16* ar = A + (size_t)arow * DIN + kc;
    Frag16 a;
    a.h[0] = *reinterpret_cast<const v8h*>(ar + ahalf);
    a.h[1] = *reinterpret_cast<const v8h*>(ar + ahalf + 8);
    if (kc + 32 < DIN) __builtin_prefetch(ar + 32, 0, 1);   // global_prefetch_b8

    const h16* wb = Wsw + ((size_t)(kc >> 5) * (DOUT >> 4) + ctile0) * 512 + lane * 16;
#pragma unroll
    for (int ct = 0; ct < 4; ++ct) {
      Frag16 b;
      b.h[0] = *reinterpret_cast<const v8h*>(wb + ct * 512);
      b.h[1] = *reinterpret_cast<const v8h*>(wb + ct * 512 + 8);
      acc[ct] = __builtin_amdgcn_wmma_f32_16x16x32_f16(
          false, a.v, false, b.v, (short)0, acc[ct], false, false);
    }
  }

  const int orow0 = r0 + (hi ? 8 : 0);   // C/D layout: lanes>=16 hold M = r + 8
#pragma unroll
  for (int ct = 0; ct < 4; ++ct) {
    int col = cg + ct * 16 + (lane & 15);
    float bv = bias[col];
    float gv = gamma[col] * BN_INV;
    float ev = beta[col];
#pragma unroll
    for (int r = 0; r < 8; ++r) {
      float v = acc[ct][r] + bv;
      v = fmaxf(v, 0.0f);
      v = fmaf(v, gv, ev);
      out[(size_t)(orow0 + r) * DOUT + col] = (h16)v;
    }
  }
}

// ---------------------------------------------------------------- max over K=3 edges
__global__ void max_edges(const h16* __restrict__ H, float* __restrict__ xf,
                          h16* __restrict__ xh) {
  int t = blockIdx.x * blockDim.x + threadIdx.x;
  if (t >= NPTS * 64) return;
  int i = t >> 6, c = t & 63;
  const h16* p = H + (size_t)i * 3 * 64 + c;
  float v = fmaxf(fmaxf((float)p[0], (float)p[64]), (float)p[128]);
  xf[t] = v;
  xh[t] = (h16)v;
}

__global__ void concat3(const h16* __restrict__ x1, const h16* __restrict__ x2,
                        const h16* __restrict__ x3, h16* __restrict__ C) {
  int t = blockIdx.x * blockDim.x + threadIdx.x;
  if (t >= NPTS * 192) return;
  int i = t / 192, c = t % 192;
  h16 v = (c < 64) ? x1[i * 64 + c]
        : (c < 128) ? x2[i * 64 + (c - 64)] : x3[i * 64 + (c - 128)];
  C[t] = v;
}

// ---------------------------------------------------------------- final linear (128->13) + log_softmax
__global__ __launch_bounds__(256) void final_logsoftmax(
    const h16* __restrict__ H, const float* __restrict__ W,
    const float* __restrict__ b, float* __restrict__ out) {
  __shared__ float Ws[128 * 13];
  __shared__ float bs[13];
  for (int k = threadIdx.x; k < 128 * 13; k += 256) Ws[k] = W[k];
  if (threadIdx.x < 13) bs[threadIdx.x] = b[threadIdx.x];
  __syncthreads();
  int i = blockIdx.x * 256 + threadIdx.x;   // grid = NPTS/256 exactly
  float h[128];
#pragma unroll
  for (int d = 0; d < 128; ++d) h[d] = (float)H[(size_t)i * 128 + d];
  float l[13];
#pragma unroll
  for (int c = 0; c < 13; ++c) {
    float acc = bs[c];
    for (int d = 0; d < 128; ++d) acc = fmaf(h[d], Ws[d * 13 + c], acc);
    l[c] = acc;
  }
  float m = l[0];
#pragma unroll
  for (int c = 1; c < 13; ++c) m = fmaxf(m, l[c]);
  float s = 0.0f;
#pragma unroll
  for (int c = 0; c < 13; ++c) s += __expf(l[c] - m);
  float lse = __logf(s);
#pragma unroll
  for (int c = 0; c < 13; ++c) out[(size_t)i * 13 + c] = l[c] - m - lse;
}

// ---------------------------------------------------------------- launcher
extern "C" void kernel_launch(void* const* d_in, const int* in_sizes, int n_in,
                              void* d_out, int out_size, void* d_ws, size_t ws_size,
                              hipStream_t stream) {
  const float* x   = (const float*)d_in[0];
  const float* pos = (const float*)d_in[1];
  // d_in[2] = batch (contiguous clouds; unused)
  // params flattened via sorted pytree keys: c1,c2,c3,fin,lin1,m1,m2; per layer W,b,be,g
  const float* c1W0 = (const float*)d_in[3];
  const float* c1b0 = (const float*)d_in[4];
  const float* c1be0= (const float*)d_in[5];
  const float* c1g0 = (const float*)d_in[6];
  const float* c1W1 = (const float*)d_in[7];
  const float* c1b1 = (const float*)d_in[8];
  const float* c1be1= (const float*)d_in[9];
  const float* c1g1 = (const float*)d_in[10];
  const float* c2W0 = (const float*)d_in[11];
  const float* c2b0 = (const float*)d_in[12];
  const float* c2be0= (const float*)d_in[13];
  const float* c2g0 = (const float*)d_in[14];
  const float* c2W1 = (const float*)d_in[15];
  const float* c2b1 = (const float*)d_in[16];
  const float* c2be1= (const float*)d_in[17];
  const float* c2g1 = (const float*)d_in[18];
  const float* c3W0 = (const float*)d_in[19];
  const float* c3b0 = (const float*)d_in[20];
  const float* c3be0= (const float*)d_in[21];
  const float* c3g0 = (const float*)d_in[22];
  const float* c3W1 = (const float*)d_in[23];
  const float* c3b1 = (const float*)d_in[24];
  const float* c3be1= (const float*)d_in[25];
  const float* c3g1 = (const float*)d_in[26];
  const float* finW = (const float*)d_in[27];
  const float* finb = (const float*)d_in[28];
  const float* l1W  = (const float*)d_in[29];
  const float* l1b  = (const float*)d_in[30];
  const float* l1be = (const float*)d_in[31];
  const float* l1g  = (const float*)d_in[32];
  const float* m1W  = (const float*)d_in[33];
  const float* m1b  = (const float*)d_in[34];
  const float* m1be = (const float*)d_in[35];
  const float* m1g  = (const float*)d_in[36];
  const float* m2W  = (const float*)d_in[37];
  const float* m2b  = (const float*)d_in[38];
  const float* m2be = (const float*)d_in[39];
  const float* m2g  = (const float*)d_in[40];

  char* p = (char*)d_ws;
  auto carve = [&](size_t bytes) -> char* {
    char* r = p;
    p += (bytes + 255) & ~(size_t)255;
    return r;
  };
  float* x0f = (float*)carve((size_t)NPTS * 6 * 4);
  h16*   x0h = (h16*)  carve((size_t)NPTS * 6 * 2);
  float* x1f = (float*)carve((size_t)NPTS * 64 * 4);
  h16*   x1h = (h16*)  carve((size_t)NPTS * 64 * 2);
  float* x2f = (float*)carve((size_t)NPTS * 64 * 4);
  h16*   x2h = (h16*)  carve((size_t)NPTS * 64 * 2);
  float* x3f = (float*)carve((size_t)NPTS * 64 * 4);
  h16*   x3h = (h16*)  carve((size_t)NPTS * 64 * 2);
  float* sq  = (float*)carve((size_t)NPTS * 4);
  int*   nbr = (int*)  carve((size_t)NPTS * 3 * 4);
  h16*   E    = (h16*)carve((size_t)NEDGE * 128 * 2);
  h16*   H1   = (h16*)carve((size_t)NEDGE * 64 * 2);
  h16*   H2   = (h16*)carve((size_t)NEDGE * 64 * 2);
  h16*   C192 = (h16*)carve((size_t)NPTS * 192 * 2);
  h16*   Hk   = (h16*)carve((size_t)NPTS * 1024 * 2);
  h16*   H256 = (h16*)carve((size_t)NPTS * 256 * 2);
  h16*   H128 = (h16*)carve((size_t)NPTS * 128 * 2);
  h16* Wc1a = (h16*)carve(32 * 64 * 2);
  h16* Wc1b = (h16*)carve(64 * 64 * 2);
  h16* Wc2a = (h16*)carve(128 * 64 * 2);
  h16* Wc2b = (h16*)carve(64 * 64 * 2);
  h16* Wc3a = (h16*)carve(128 * 64 * 2);
  h16* Wc3b = (h16*)carve(64 * 64 * 2);
  h16* Wl1  = (h16*)carve(192 * 1024 * 2);
  h16* Wm1  = (h16*)carve(1024 * 256 * 2);
  h16* Wm2  = (h16*)carve(256 * 128 * 2);

  auto cdiv = [](int a, int b) { return (a + b - 1) / b; };
  const int T = 256;

  build_x0<<<cdiv(NPTS * 6, T), T, 0, stream>>>(x, pos, x0f, x0h);

  convert_swizzle_b<<<cdiv(32 * 64, T),    T, 0, stream>>>(c1W0, Wc1a, 12,   32,   64);
  convert_swizzle_b<<<cdiv(64 * 64, T),    T, 0, stream>>>(c1W1, Wc1b, 64,   64,   64);
  convert_swizzle_b<<<cdiv(128 * 64, T),   T, 0, stream>>>(c2W0, Wc2a, 128,  128,  64);
  convert_swizzle_b<<<cdiv(64 * 64, T),    T, 0, stream>>>(c2W1, Wc2b, 64,   64,   64);
  convert_swizzle_b<<<cdiv(128 * 64, T),   T, 0, stream>>>(c3W0, Wc3a, 128,  128,  64);
  convert_swizzle_b<<<cdiv(64 * 64, T),    T, 0, stream>>>(c3W1, Wc3b, 64,   64,   64);
  convert_swizzle_b<<<cdiv(192 * 1024, T), T, 0, stream>>>(l1W,  Wl1,  192,  192,  1024);
  convert_swizzle_b<<<cdiv(1024 * 256, T), T, 0, stream>>>(m1W,  Wm1,  1024, 1024, 256);
  convert_swizzle_b<<<cdiv(256 * 128, T),  T, 0, stream>>>(m2W,  Wm2,  256,  256,  128);

  // ---- EdgeConv 1 (D=6, din=12 padded to 32)
  row_norms<6><<<cdiv(NPTS, T), T, 0, stream>>>(x0f, sq);
  knn_kernel<6><<<NPTS / T, T, 0, stream>>>(x0f, sq, nbr);
  edge_gather<6, 32><<<cdiv(NEDGE, T), T, 0, stream>>>(x0h, nbr, E);
  gemm_bn_relu<32, 64><<<dim3(NEDGE / 128, 1), T, 0, stream>>>(E, Wc1a, c1b0, c1g0, c1be0, H1);
  gemm_bn_relu<64, 64><<<dim3(NEDGE / 128, 1), T, 0, stream>>>(H1, Wc1b, c1b1, c1g1, c1be1, H2);
  max_edges<<<cdiv(NPTS * 64, T), T, 0, stream>>>(H2, x1f, x1h);

  // ---- EdgeConv 2 (D=64, din=128)
  row_norms<64><<<cdiv(NPTS, T), T, 0, stream>>>(x1f, sq);
  knn_kernel<64><<<NPTS / T, T, 0, stream>>>(x1f, sq, nbr);
  edge_gather<64, 128><<<cdiv(NEDGE, T), T, 0, stream>>>(x1h, nbr, E);
  gemm_bn_relu<128, 64><<<dim3(NEDGE / 128, 1), T, 0, stream>>>(E, Wc2a, c2b0, c2g0, c2be0, H1);
  gemm_bn_relu<64, 64><<<dim3(NEDGE / 128, 1), T, 0, stream>>>(H1, Wc2b, c2b1, c2g1, c2be1, H2);
  max_edges<<<cdiv(NPTS * 64, T), T, 0, stream>>>(H2, x2f, x2h);

  // ---- EdgeConv 3 (D=64, din=128)
  row_norms<64><<<cdiv(NPTS, T), T, 0, stream>>>(x2f, sq);
  knn_kernel<64><<<NPTS / T, T, 0, stream>>>(x2f, sq, nbr);
  edge_gather<64, 128><<<cdiv(NEDGE, T), T, 0, stream>>>(x2h, nbr, E);
  gemm_bn_relu<128, 64><<<dim3(NEDGE / 128, 1), T, 0, stream>>>(E, Wc3a, c3b0, c3g0, c3be0, H1);
  gemm_bn_relu<64, 64><<<dim3(NEDGE / 128, 1), T, 0, stream>>>(H1, Wc3b, c3b1, c3g1, c3be1, H2);
  max_edges<<<cdiv(NPTS * 64, T), T, 0, stream>>>(H2, x3f, x3h);

  // ---- head MLPs
  concat3<<<cdiv(NPTS * 192, T), T, 0, stream>>>(x1h, x2h, x3h, C192);
  gemm_bn_relu<192, 1024><<<dim3(NPTS / 128, 16), T, 0, stream>>>(C192, Wl1, l1b, l1g, l1be, Hk);
  gemm_bn_relu<1024, 256><<<dim3(NPTS / 128, 4),  T, 0, stream>>>(Hk, Wm1, m1b, m1g, m1be, H256);
  gemm_bn_relu<256, 128><<<dim3(NPTS / 128, 2),   T, 0, stream>>>(H256, Wm2, m2b, m2g, m2be, H128);
  final_logsoftmax<<<NPTS / T, T, 0, stream>>>(H128, finW, finb, (float*)d_out);
}
